// GINTopK_65712999628885
// MI455X (gfx1250) — matrix operand order repaired and checked by reading, not compiled, not run on hardware
//
#include <hip/hip_runtime.h>
#include <hip/hip_bf16.h>

// ---------------- problem constants ----------------
constexpr int NB   = 8;        // batches
constexpr int NN0  = 16384;    // initial nodes per graph
constexpr int NHW  = 128;      // hidden width (== IW)
constexpr int NEG  = 98304;    // edges per graph
constexpr int NDEP = 4;        // depth

typedef __attribute__((ext_vector_type(16))) _Float16 v16h;
typedef __attribute__((ext_vector_type(8)))  _Float16 v8h;
typedef __attribute__((ext_vector_type(8)))  float    v8f;

// ---------------- small utilities ----------------
__global__ void zero_f32_kernel(float* p, int n) {
    int i = blockIdx.x * blockDim.x + threadIdx.x;
    if (i < n) p[i] = 0.0f;
}
__global__ void zero_i32_kernel(int* p, int n) {
    int i = blockIdx.x * blockDim.x + threadIdx.x;
    if (i < n) p[i] = 0;
}

__global__ void init_edges_kernel(const int* __restrict__ src, const int* __restrict__ dst,
                                  int* __restrict__ esrc, int* __restrict__ edst,
                                  int* __restrict__ evalid) {
    int t = blockIdx.x * blockDim.x + threadIdx.x;
    if (t >= NB * NEG) return;
    esrc[t] = src[t];
    edst[t] = dst[t];
    evalid[t] = 1;
}

// copy rows [0,n) of x (stride NN0 rows) into h  (float4 vectorized)
__global__ void copy_rows_kernel(const float* __restrict__ x, float* __restrict__ h, int n) {
    long long t = (long long)blockIdx.x * blockDim.x + threadIdx.x;
    long long total = (long long)NB * n * (NHW / 4);
    if (t >= total) return;
    int perB = n * (NHW / 4);
    int b   = (int)(t / perB);
    int r   = (int)(t % perB);
    int row = r / (NHW / 4);
    int c4  = r % (NHW / 4);
    size_t off = ((size_t)b * NN0 + row) * NHW;
    ((float4*)(h + off))[c4] = ((const float4*)(x + off))[c4];
}

// h[dst] += x[src] for every valid edge; 32 lanes/edge, 4 channels/lane
__global__ void scatter_kernel(const float* __restrict__ x, float* __restrict__ h,
                               const int* __restrict__ esrc, const int* __restrict__ edst,
                               const int* __restrict__ evalid) {
    long long tid = (long long)blockIdx.x * blockDim.x + threadIdx.x;
    long long total = (long long)NB * NEG * 32;
    if (tid >= total) return;
    int lane = (int)(tid & 31);
    long long e = tid >> 5;
    int b = (int)(e / NEG);
    int j = (int)(e % NEG);
    int eidx = b * NEG + j;
    if (!evalid[eidx]) return;
    int s = esrc[eidx], d = edst[eidx];
    const float* xs = x + ((size_t)b * NN0 + s) * NHW + lane * 4;
    float*       hd = h + ((size_t)b * NN0 + d) * NHW + lane * 4;
#pragma unroll
    for (int c = 0; c < 4; ++c) atomicAdd(hd + c, xs[c]);
}

// f32 rows [0,n) -> f16 (same [B][NN0][NHW] layout)
__global__ void convert_f16_kernel(const float* __restrict__ in, _Float16* __restrict__ out, int n) {
    long long t = (long long)blockIdx.x * blockDim.x + threadIdx.x;
    long long total = (long long)NB * n * (NHW / 4);
    if (t >= total) return;
    int perB = n * (NHW / 4);
    int b   = (int)(t / perB);
    int r   = (int)(t % perB);
    int row = r / (NHW / 4);
    int c4  = r % (NHW / 4);
    size_t off = ((size_t)b * NN0 + row) * NHW + (size_t)c4 * 4;
    float4 v = *(const float4*)(in + off);
    _Float16* o = out + off;
    o[0] = (_Float16)v.x; o[1] = (_Float16)v.y; o[2] = (_Float16)v.z; o[3] = (_Float16)v.w;
}

// transpose-convert weight W[K][N] (128x128 f32) -> Wt[N][K] f16
__global__ void wconv_kernel(const float* __restrict__ w, _Float16* __restrict__ wt) {
    int t = blockIdx.x * blockDim.x + threadIdx.x;
    if (t >= NHW * NHW) return;
    int n = t / NHW, k = t % NHW;
    wt[t] = (_Float16)w[k * NHW + n];
}

// ---------------- WMMA GEMM: out = relu(A[n,128] @ W[128,128] + bias) ----------------
// A: f16 row-major, row stride NHW, batch stride NN0*NHW. Wt: [N][K] f16 (transposed).
// Whole 32KB weight + bias staged in LDS once per block (8 waves share it); each
// wave computes a 16-row strip x all 128 cols (8 n-tiles), K=128 in 4 WMMA steps,
// with all 4 B-fragments hoisted so the v_wmma chain issues back-to-back.
template <bool F16OUT>
__global__ void gemm_relu_kernel(const _Float16* __restrict__ A, const _Float16* __restrict__ Wt,
                                 const float* __restrict__ bias, void* __restrict__ Out,
                                 int rowTiles) {
    __shared__ _Float16 sW[NHW * NHW];   // 32 KB transposed weights
    __shared__ float    sB[NHW];         // bias
    {
        const float4* srcW = (const float4*)Wt;
        float4*       dstW = (float4*)sW;
        for (int i = threadIdx.x; i < (NHW * NHW) / 8; i += blockDim.x) dstW[i] = srcW[i];
        for (int i = threadIdx.x; i < NHW; i += blockDim.x) sB[i] = bias[i];
    }
    __syncthreads();

    int wave = threadIdx.x >> 5;
    int lane = threadIdx.x & 31;
    int tile = blockIdx.x * (blockDim.x >> 5) + wave;
    if (tile >= rowTiles) return;            // uniform per wave -> EXEC stays all-ones
    int b = blockIdx.y;
    size_t base = ((size_t)b * NN0 + (size_t)tile * 16) * NHW;

    // A fragment layout (16-bit A 16x32): lane L: row M = L%16;
    // kb = (L<16)?0:8 ; elems 0..7 -> K=kb..kb+7, elems 8..15 -> K=16+kb..16+kb+7
    int mA  = lane & 15;
    int kbA = (lane < 16) ? 0 : 8;
    // B fragment layout (32x16): lane L: col N = L%16; kb = (L<16)?0:8; elems 0..7 -> K=kb..kb+7, elems 8..15 -> K=16+kb..16+kb+7
    int nB  = lane & 15;
    int kbB = (lane < 16) ? 0 : 8;

    const _Float16* arow = A + base + (size_t)mA * NHW;
    // prefetch the next row strip of streamed activations handled by this block slot
    __builtin_prefetch(arow + (size_t)(gridDim.x) * (blockDim.x >> 5) * 16 * NHW, 0, 1);
    v16h afrag[4];
#pragma unroll
    for (int ks = 0; ks < 4; ++ks) {
        int k0 = ks * 32;
        v8h lo = *(const v8h*)(arow + k0 + kbA);
        v8h hi = *(const v8h*)(arow + k0 + 16 + kbA);
        v16h a;
#pragma unroll
        for (int i = 0; i < 8; ++i) { a[i] = lo[i]; a[i + 8] = hi[i]; }
        afrag[ks] = a;
    }

    int rowOff = (lane < 16) ? 0 : 8;   // C/D layout: VGPR r -> M = r (+8 for high lanes)
#pragma unroll
    for (int nt = 0; nt < 8; ++nt) {
        int col = nt * 16 + nB;
        const _Float16* wrow = sW + (size_t)col * NHW + kbB;
        // hoist all 4 K-step B fragments from LDS so the 4 WMMAs chain
        v16h bfrag[4];
#pragma unroll
        for (int ks = 0; ks < 4; ++ks) {
            v8h lo = *(const v8h*)(wrow + ks * 32);
            v8h hi = *(const v8h*)(wrow + ks * 32 + 16);
            v16h bf;
#pragma unroll
            for (int i = 0; i < 8; ++i) { bf[i] = lo[i]; bf[i + 8] = hi[i]; }
            bfrag[ks] = bf;
        }
        float bv = sB[col];
        v8f acc;
#pragma unroll
        for (int r = 0; r < 8; ++r) acc[r] = bv;   // column fixed per lane -> bias preload
#pragma unroll
        for (int ks = 0; ks < 4; ++ks) {
            acc = __builtin_amdgcn_wmma_f32_16x16x32_f16(
                false, afrag[ks], false, bfrag[ks], (short)0, acc, false, false);
        }
#pragma unroll
        for (int r = 0; r < 8; ++r) {
            float v = acc[r] > 0.0f ? acc[r] : 0.0f;
            size_t idx = base + (size_t)(rowOff + r) * NHW + col;
            if (F16OUT) ((_Float16*)Out)[idx] = (_Float16)v;
            else        ((float*)Out)[idx]    = v;
        }
    }
}

// ---------------- pooling: score, radix-select, compact, gather ----------------
__global__ void norm_kernel(const float* __restrict__ pw, float* __restrict__ invnorm) {
    __shared__ float s[NHW];
    int t = threadIdx.x;
    s[t] = pw[t] * pw[t];
    __syncthreads();
    for (int o = NHW / 2; o > 0; o >>= 1) {
        if (t < o) s[t] += s[t + o];
        __syncthreads();
    }
    if (t == 0) invnorm[0] = rsqrtf(s[0]);
}

// warp per node: score = tanh(dot(x,pw)*invnorm); key = order-preserving uint
__global__ void score_kernel(const float* __restrict__ x, const float* __restrict__ pw,
                             const float* __restrict__ invnorm,
                             float* __restrict__ score, unsigned* __restrict__ keys, int n) {
    int gw = (blockIdx.x * blockDim.x + threadIdx.x) / 32;
    int lane = threadIdx.x & 31;
    int b = blockIdx.y;
    if (gw >= n) return;
    const float* row = x + ((size_t)b * NN0 + gw) * NHW;
    float acc = 0.0f;
#pragma unroll
    for (int c = lane; c < NHW; c += 32) acc += row[c] * pw[c];
#pragma unroll
    for (int o = 16; o > 0; o >>= 1) acc += __shfl_xor(acc, o, 32);
    if (lane == 0) {
        float sc = tanhf(acc * invnorm[0]);
        score[b * NN0 + gw] = sc;
        unsigned fb = __float_as_uint(sc);
        keys[b * NN0 + gw] = (fb & 0x80000000u) ? ~fb : (fb | 0x80000000u);
    }
}

// one block per batch; MSB-first radix select of the k-th largest key
__global__ void radix_pass_kernel(const unsigned* __restrict__ keys, int n, int pass, int k,
                                  unsigned* __restrict__ prefix, int* __restrict__ kremain) {
    int b = blockIdx.x;
    __shared__ unsigned hist[256];
    for (int i = threadIdx.x; i < 256; i += blockDim.x) hist[i] = 0;
    __syncthreads();
    int shift = 8 * (3 - pass);
    unsigned pfx  = (pass == 0) ? 0u : prefix[b];
    unsigned mask = (pass == 0) ? 0u : (0xFFFFFFFFu << (shift + 8));
    for (int i = threadIdx.x; i < n; i += blockDim.x) {
        unsigned key = keys[b * NN0 + i];
        if ((key & mask) == (pfx & mask)) atomicAdd(&hist[(key >> shift) & 255], 1u);
    }
    __syncthreads();
    if (threadIdx.x == 0) {
        int kr = (pass == 0) ? k : kremain[b];
        unsigned cum = 0;
        int v = 255;
        for (; v > 0; --v) {
            if (cum + hist[v] >= (unsigned)kr) break;
            cum += hist[v];
        }
        prefix[b]  = pfx | ((unsigned)v << shift);
        kremain[b] = kr - (int)cum;
    }
}

// assign new ids: all keys > T, plus (kremain) keys == T
__global__ void compact_kernel(const unsigned* __restrict__ keys,
                               const unsigned* __restrict__ prefix,
                               const int* __restrict__ kremain, int n, int k,
                               int* __restrict__ mapping, int* __restrict__ sel_old,
                               int* __restrict__ cnt) {
    int i = blockIdx.x * blockDim.x + threadIdx.x;
    int b = blockIdx.y;
    if (i >= n) return;
    unsigned key = keys[b * NN0 + i];
    unsigned T = prefix[b];
    int keq = kremain[b];
    int pos = -1;
    if (key > T) {
        pos = atomicAdd(&cnt[2 * b], 1);
    } else if (key == T) {
        int q = atomicAdd(&cnt[2 * b + 1], 1);
        if (q < keq) pos = (k - keq) + q;
    }
    mapping[b * NN0 + i] = pos;
    if (pos >= 0) sel_old[b * NN0 + pos] = i;
}

// warp per new row: xout[pos] = h[old] * score[old]
__global__ void gather_kernel(const float* __restrict__ h, const float* __restrict__ score,
                              const int* __restrict__ sel_old, float* __restrict__ xout, int k) {
    int gw = (blockIdx.x * blockDim.x + threadIdx.x) / 32;
    int lane = threadIdx.x & 31;
    int b = blockIdx.y;
    if (gw >= k) return;
    int i = sel_old[b * NN0 + gw];
    float val = score[b * NN0 + i];
    float4 v = ((const float4*)(h + ((size_t)b * NN0 + i) * NHW))[lane];
    v.x *= val; v.y *= val; v.z *= val; v.w *= val;
    ((float4*)(xout + ((size_t)b * NN0 + gw) * NHW))[lane] = v;
}

__global__ void remap_edges_kernel(const int* __restrict__ mapping,
                                   int* __restrict__ esrc, int* __restrict__ edst,
                                   int* __restrict__ evalid) {
    int t = blockIdx.x * blockDim.x + threadIdx.x;
    if (t >= NB * NEG) return;
    if (!evalid[t]) { esrc[t] = 0; edst[t] = 0; return; }
    int b = t / NEG;
    int ns = mapping[b * NN0 + esrc[t]];
    int nd = mapping[b * NN0 + edst[t]];
    int ok = (ns >= 0) && (nd >= 0);
    esrc[t] = ok ? ns : 0;
    edst[t] = ok ? nd : 0;
    evalid[t] = ok;
}

// racc[b][c] += max_i x[b][i][c] ; racc[b][128+c] += mean_i x[b][i][c]
__global__ void readout_kernel(const float* __restrict__ x, float* __restrict__ racc, int rows) {
    int b = blockIdx.x;
    int c = threadIdx.x;      // 128 threads, coalesced across channels
    float mx = -3.402823466e+38f, sm = 0.0f;
    const float* base = x + (size_t)b * NN0 * NHW + c;
    for (int i = 0; i < rows; ++i) {
        float v = base[(size_t)i * NHW];
        mx = fmaxf(mx, v);
        sm += v;
    }
    racc[b * 2 * NHW + c]       += mx;
    racc[b * 2 * NHW + NHW + c] += sm / (float)rows;
}

// ---------------- predictor head: 256 -> 128 -> 64 -> 2 ----------------
__global__ void mlp_kernel(const float* __restrict__ racc,
                           const float* __restrict__ q1w, const float* __restrict__ q1b,
                           const float* __restrict__ q2w, const float* __restrict__ q2b,
                           const float* __restrict__ q3w, const float* __restrict__ q3b,
                           float* __restrict__ out) {
    __shared__ float r[NB * 256];
    __shared__ float h1[NB * 128];
    __shared__ float h2[NB * 64];
    int t = threadIdx.x;
    for (int i = t; i < NB * 256; i += blockDim.x) r[i] = racc[i];
    __syncthreads();
    for (int i = t; i < NB * 128; i += blockDim.x) {
        int b = i / 128, o = i % 128;
        float a = q1b[o];
        for (int j = 0; j < 256; ++j) a += r[b * 256 + j] * q1w[j * 128 + o];
        h1[i] = a > 0.0f ? a : 0.0f;
    }
    __syncthreads();
    for (int i = t; i < NB * 64; i += blockDim.x) {
        int b = i / 64, o = i % 64;
        float a = q2b[o];
        for (int j = 0; j < 128; ++j) a += h1[b * 128 + j] * q2w[j * 64 + o];
        h2[i] = a > 0.0f ? a : 0.0f;
    }
    __syncthreads();
    for (int i = t; i < NB * 2; i += blockDim.x) {
        int b = i / 2, o = i % 2;
        float a = q3b[o];
        for (int j = 0; j < 64; ++j) a += h2[b * 64 + j] * q3w[j * 2 + o];
        out[i] = a;
    }
}

// ---------------- host orchestration ----------------
extern "C" void kernel_launch(void* const* d_in, const int* in_sizes, int n_in,
                              void* d_out, int out_size, void* d_ws, size_t ws_size,
                              hipStream_t stream) {
    (void)in_sizes; (void)n_in; (void)out_size; (void)ws_size;

    const float* x_in = (const float*)d_in[0];
    const int*   src  = (const int*)d_in[1];
    const int*   dst  = (const int*)d_in[2];
    // layer params: d_in[3+5i .. 3+5i+4] = w1, b1, w2, b2, pw
    const float* q1w = (const float*)d_in[23];
    const float* q1b = (const float*)d_in[24];
    const float* q2w = (const float*)d_in[25];
    const float* q2b = (const float*)d_in[26];
    const float* q3w = (const float*)d_in[27];
    const float* q3b = (const float*)d_in[28];
    float* out = (float*)d_out;

    // ---- workspace bump allocator (256B aligned) ----
    char* p = (char*)d_ws;
    auto alloc = [&](size_t bytes) -> void* {
        void* r = (void*)p;
        p += (bytes + 255) & ~(size_t)255;
        return r;
    };
    const size_t NODE_F = (size_t)NB * NN0 * NHW;
    float*     bufH   = (float*)alloc(NODE_F * sizeof(float));     // agg / post-GEMM2 x
    float*     bufX   = (float*)alloc(NODE_F * sizeof(float));     // pooled x
    _Float16*  f16a   = (_Float16*)alloc(NODE_F * sizeof(_Float16));
    _Float16*  f16b   = (_Float16*)alloc(NODE_F * sizeof(_Float16));
    _Float16*  wt1    = (_Float16*)alloc((size_t)NHW * NHW * sizeof(_Float16));
    _Float16*  wt2    = (_Float16*)alloc((size_t)NHW * NHW * sizeof(_Float16));
    float*     score  = (float*)alloc((size_t)NB * NN0 * sizeof(float));
    unsigned*  keys   = (unsigned*)alloc((size_t)NB * NN0 * sizeof(unsigned));
    int*       mapping= (int*)alloc((size_t)NB * NN0 * sizeof(int));
    int*       selold = (int*)alloc((size_t)NB * NN0 * sizeof(int));
    int*       esrc   = (int*)alloc((size_t)NB * NEG * sizeof(int));
    int*       edst   = (int*)alloc((size_t)NB * NEG * sizeof(int));
    int*       evalid = (int*)alloc((size_t)NB * NEG * sizeof(int));
    float*     racc   = (float*)alloc((size_t)NB * 2 * NHW * sizeof(float));
    float*     invnrm = (float*)alloc(256);
    unsigned*  prefix = (unsigned*)alloc((size_t)NB * sizeof(unsigned));
    int*       kremain= (int*)alloc((size_t)NB * sizeof(int));
    int*       cnt    = (int*)alloc((size_t)2 * NB * sizeof(int));

    // ---- init ----
    zero_f32_kernel<<<(NB * 2 * NHW + 255) / 256, 256, 0, stream>>>(racc, NB * 2 * NHW);
    init_edges_kernel<<<(NB * NEG + 255) / 256, 256, 0, stream>>>(src, dst, esrc, edst, evalid);

    const float* curX = x_in;
    int n = NN0;
    for (int layer = 0; layer < NDEP; ++layer) {
        const float* w1 = (const float*)d_in[3 + 5 * layer + 0];
        const float* b1 = (const float*)d_in[3 + 5 * layer + 1];
        const float* w2 = (const float*)d_in[3 + 5 * layer + 2];
        const float* b2 = (const float*)d_in[3 + 5 * layer + 3];
        const float* pw = (const float*)d_in[3 + 5 * layer + 4];
        int k = (4 * n + 4) / 5;                 // ceil(0.8*n)
        int rowsPad  = (n + 15) & ~15;
        int rowTiles = rowsPad / 16;

        // GIN aggregation: h = x + scatter_add(x[src] -> dst)
        {
            long long tot4 = (long long)NB * n * (NHW / 4);
            copy_rows_kernel<<<(unsigned)((tot4 + 255) / 256), 256, 0, stream>>>(curX, bufH, n);
            long long te = (long long)NB * NEG * 32;
            scatter_kernel<<<(unsigned)((te + 255) / 256), 256, 0, stream>>>(curX, bufH, esrc, edst, evalid);
        }

        // MLP: x = relu(relu(h@w1+b1)@w2+b2)  via WMMA f16->f32
        {
            long long tot4 = (long long)NB * n * (NHW / 4);
            convert_f16_kernel<<<(unsigned)((tot4 + 255) / 256), 256, 0, stream>>>(bufH, f16a, n);
            wconv_kernel<<<(NHW * NHW + 255) / 256, 256, 0, stream>>>(w1, wt1);
            wconv_kernel<<<(NHW * NHW + 255) / 256, 256, 0, stream>>>(w2, wt2);
            dim3 grid((rowTiles + 7) / 8, NB);
            gemm_relu_kernel<true><<<grid, 256, 0, stream>>>(f16a, wt1, b1, (void*)f16b, rowTiles);
            gemm_relu_kernel<false><<<grid, 256, 0, stream>>>(f16b, wt2, b2, (void*)bufH, rowTiles);
        }

        // TopK pooling
        norm_kernel<<<1, NHW, 0, stream>>>(pw, invnrm);
        {
            dim3 grid((unsigned)(((long long)n * 32 + 255) / 256), NB);
            score_kernel<<<grid, 256, 0, stream>>>(bufH, pw, invnrm, score, keys, n);
        }
        for (int pass = 0; pass < 4; ++pass)
            radix_pass_kernel<<<NB, 256, 0, stream>>>(keys, n, pass, k, prefix, kremain);
        zero_i32_kernel<<<1, 64, 0, stream>>>(cnt, 2 * NB);
        {
            dim3 grid((n + 255) / 256, NB);
            compact_kernel<<<grid, 256, 0, stream>>>(keys, prefix, kremain, n, k, mapping, selold, cnt);
        }
        {
            dim3 grid((unsigned)(((long long)k * 32 + 255) / 256), NB);
            gather_kernel<<<grid, 256, 0, stream>>>(bufH, score, selold, bufX, k);
        }
        remap_edges_kernel<<<(NB * NEG + 255) / 256, 256, 0, stream>>>(mapping, esrc, edst, evalid);

        // readout on pooled x
        readout_kernel<<<NB, NHW, 0, stream>>>(bufX, racc, k);

        curX = bufX;
        n = k;
    }

    mlp_kernel<<<1, 256, 0, stream>>>(racc, q1w, q1b, q2w, q2b, q3w, q3b, out);
}
